// GNNReID_39633958208209
// MI455X (gfx1250) — matrix operand order repaired and checked by reading, not compiled, not run on hardware
//
#include <hip/hip_runtime.h>

// ---------------------------------------------------------------------------
// GNN transformer forward for MI455X (gfx1250, wave32, WMMA).
//
// Roofline: ~840 GFLOP of dense GEMM vs ~400 MB unique weight bytes
// => ~2000 FLOP/B, far above the 23.3 TB/s HBM knee, so the run is
// matrix-pipe bound.  v_wmma_f32_16x16x32_bf16 moves 8x the FLOPs per
// instruction of the f32 WMMA, so weights/activations are converted to
// bf16 on the fly (f32 accumulate).  The B operand is transposed during
// the f32->bf16 conversion so both WMMA fragments are contiguous LDS
// reads matching the ISA 7.12.2 VGPR layouts.
//
// GEMM staging is double-buffered in LDS using the gfx1250 async
// global->LDS ops (ASYNCcnt + s_wait_asynccnt) when available, else a
// software-pipelined VGPR staging path.  Ragged-N B-tile rows are clamped
// (not zero-filled): out-of-range columns are never stored.
// ---------------------------------------------------------------------------

typedef __attribute__((ext_vector_type(16))) __bf16 v16bf;
typedef __attribute__((ext_vector_type(8)))  __bf16 v8bf;
typedef __attribute__((ext_vector_type(4)))  __bf16 v4bf;
typedef __attribute__((ext_vector_type(8)))  float  v8f;
typedef int v4i_ __attribute__((vector_size(16)));   // matches builtin param

#define AS1 __attribute__((address_space(1)))
#define AS3 __attribute__((address_space(3)))

namespace {
constexpr int   N_   = 4096;
constexpr int   E_   = 65536;
constexpr int   D_   = 2048;
constexpr int   H_   = 8;
constexpr int   L_   = 2;
constexpr int   DH_  = D_ / H_;    // 256
constexpr int   HID_ = 4 * D_;     // 8192
constexpr int   NC_  = 751;
constexpr float SCALE_ = 0.0625f;  // 1/sqrt(DH)
}

#if __has_builtin(__builtin_amdgcn_global_load_async_to_lds_b128)
#define USE_ASYNC_LDS 1
#else
#define USE_ASYNC_LDS 0
#endif

#if USE_ASYNC_LDS
__device__ __forceinline__ void async_ld128(const void* g, void* l) {
  // GLOBAL_LOAD_ASYNC_TO_LDS_B128: (global src, lds dst, imm offset, cpol)
  __builtin_amdgcn_global_load_async_to_lds_b128(
      (AS1 v4i_*)g, (AS3 v4i_*)l, 0, 0);
}
#endif

__device__ __forceinline__ void wait_async0() {
#if __has_builtin(__builtin_amdgcn_s_wait_asynccnt)
  __builtin_amdgcn_s_wait_asynccnt(0);
#else
  asm volatile("s_wait_asynccnt 0" ::: "memory");
#endif
}

// Order-preserving f32 <-> u32 encoding so segment-max can use atomicMax(u32).
__device__ __forceinline__ unsigned encf(float f) {
  const unsigned u = __float_as_uint(f);
  return (u & 0x80000000u) ? ~u : (u | 0x80000000u);
}
__device__ __forceinline__ float decf(unsigned e) {
  return (e & 0x80000000u) ? __uint_as_float(e ^ 0x80000000u)
                           : __uint_as_float(~e);
}

// ---------------------------------------------------------------------------
// f32 -> bf16 elementwise convert (4 elems / thread, packed 8B stores)
// ---------------------------------------------------------------------------
__global__ __launch_bounds__(256)
void cvt_f32_bf16(const float4* __restrict__ in, __bf16* __restrict__ out, int n4) {
  const int i = blockIdx.x * 256 + threadIdx.x;
  if (i >= n4) return;
  const float4 f = in[i];
  v4bf o;
  o[0] = (__bf16)f.x; o[1] = (__bf16)f.y; o[2] = (__bf16)f.z; o[3] = (__bf16)f.w;
  *(v4bf*)(out + 4 * (size_t)i) = o;
}

// ---------------------------------------------------------------------------
// W (K x Nc, f32 row-major) -> B^T (Nc x K, bf16 row-major) via 32x32 LDS tile
// ---------------------------------------------------------------------------
__global__ __launch_bounds__(256)
void transpose_cvt(const float* __restrict__ W, __bf16* __restrict__ BT,
                   int K, int Nc) {
  __shared__ float tile[32][33];
  const int tx = threadIdx.x;          // 0..31
  const int ty = threadIdx.y;          // 0..7
  const int ncol = blockIdx.x * 32 + tx;
#pragma unroll
  for (int s = 0; s < 4; ++s) {
    const int krow = blockIdx.y * 32 + ty + 8 * s;   // K is a multiple of 32
    tile[ty + 8 * s][tx] = (ncol < Nc) ? W[(size_t)krow * Nc + ncol] : 0.f;
  }
  __syncthreads();
#pragma unroll
  for (int s = 0; s < 4; ++s) {
    const int nr = blockIdx.x * 32 + ty + 8 * s;
    if (nr < Nc)
      BT[(size_t)nr * K + blockIdx.y * 32 + tx] = (__bf16)tile[tx][ty + 8 * s];
  }
}

// ---------------------------------------------------------------------------
// C = A(MxK,bf16) @ B(KxNc) + bias,  B given as B^T (Nc x K, bf16).
// 128x128 block tile, 8 waves (2x4), wave tile 64x32 = 8 v_wmma accumulators.
// Double-buffered LDS, one barrier per K-step.
// ---------------------------------------------------------------------------
template <bool OUT_BF16, bool RELU>
__global__ __launch_bounds__(256)
void gemm_wmma(const __bf16* __restrict__ A, const __bf16* __restrict__ BT,
               const float* __restrict__ bias,
               float* __restrict__ Cf, __bf16* __restrict__ Cb,
               int M, int Nc, int K) {
  __shared__ __align__(32) __bf16 sA[2][128 * 32];
  __shared__ __align__(32) __bf16 sB[2][128 * 32];

  const int tid  = threadIdx.x;
  const int lane = tid & 31;
  const int wid  = tid >> 5;
  const int wr   = wid >> 2;           // wave row  (2, 64 rows each)
  const int wc   = wid & 3;            // wave col  (4, 32 cols each)
  const int h    = lane >> 4;          // lane half (WMMA layout)
  const int lm   = lane & 15;
  const int bm   = blockIdx.y * 128;
  const int bn   = blockIdx.x * 128;

  const int r0 = tid >> 2;             // staging row  0..63
  const int c0 = (tid & 3) * 8;        // staging col  {0,8,16,24} bf16

  // Clamp ragged B rows: cols >= Nc are never stored, values can be garbage,
  // only the address must stay in-bounds.
  const int brA = (bn + r0      < Nc) ? (bn + r0)      : (Nc - 1);
  const int brB = (bn + r0 + 64 < Nc) ? (bn + r0 + 64) : (Nc - 1);
  const __bf16* __restrict__ pa0 = A  + (size_t)(bm + r0)      * K + c0;
  const __bf16* __restrict__ pa1 = A  + (size_t)(bm + r0 + 64) * K + c0;
  const __bf16* __restrict__ pb0 = BT + (size_t)brA            * K + c0;
  const __bf16* __restrict__ pb1 = BT + (size_t)brB            * K + c0;
  const int la0 = r0 * 32 + c0;
  const int la1 = (r0 + 64) * 32 + c0;

  v8f acc[4][2] = {};
  int buf = 0;

#if USE_ASYNC_LDS
  auto stage = [&](int kk, int b) {
    async_ld128(pa0 + kk, &sA[b][la0]);
    async_ld128(pa1 + kk, &sA[b][la1]);
    async_ld128(pb0 + kk, &sB[b][la0]);
    async_ld128(pb1 + kk, &sB[b][la1]);
  };
  stage(0, 0);
  wait_async0();
  __syncthreads();
#else
  {
    const uint4 a0 = *(const uint4*)pa0;
    const uint4 a1 = *(const uint4*)pa1;
    const uint4 b0 = *(const uint4*)pb0;
    const uint4 b1 = *(const uint4*)pb1;
    *(uint4*)&sA[0][la0] = a0;
    *(uint4*)&sA[0][la1] = a1;
    *(uint4*)&sB[0][la0] = b0;
    *(uint4*)&sB[0][la1] = b1;
  }
  __syncthreads();
#endif

  for (int k0 = 0; k0 < K; k0 += 32) {
    const bool hasNext = (k0 + 32) < K;

#if USE_ASYNC_LDS
    if (hasNext) stage(k0 + 32, buf ^ 1);   // stream next tile while computing
#else
    uint4 a0 = {}, a1 = {}, b0 = {}, b1 = {};
    if (hasNext) {                          // software pipeline: global->VGPR
      a0 = *(const uint4*)(pa0 + k0 + 32);
      a1 = *(const uint4*)(pa1 + k0 + 32);
      b0 = *(const uint4*)(pb0 + k0 + 32);
      b1 = *(const uint4*)(pb1 + k0 + 32);
      if (k0 + 64 < K) {
        __builtin_prefetch(pa0 + k0 + 64, 0, 1);
        __builtin_prefetch(pb0 + k0 + 64, 0, 1);
      }
    }
#endif

    // ---- fragments per ISA 7.12.2 16-bit layouts + 8 WMMAs ----
    {
      const __bf16* __restrict__ sAb = &sA[buf][0];
      const __bf16* __restrict__ sBb = &sB[buf][0];
      v16bf bfr[2];
#pragma unroll
      for (int j = 0; j < 2; ++j) {
        const int n = wc * 32 + j * 16 + lm;
        const v8bf blo = *(const v8bf*)&sBb[n * 32 + h * 8];      // K h*8..+7
        const v8bf bhi = *(const v8bf*)&sBb[n * 32 + 16 + h * 8]; // K 16+h*8..+7
        v16bf bv;
#pragma unroll
        for (int t = 0; t < 8; ++t) { bv[t] = blo[t]; bv[t + 8] = bhi[t]; }
        bfr[j] = bv;
      }
      v16bf afr[4];
#pragma unroll
      for (int i = 0; i < 4; ++i) {
        const int m = wr * 64 + i * 16 + lm;
        const v8bf lo = *(const v8bf*)&sAb[m * 32 + h * 8];      // K h*8..+7
        const v8bf hi = *(const v8bf*)&sAb[m * 32 + 16 + h * 8]; // K 16+h*8..+7
        v16bf a;
#pragma unroll
        for (int t = 0; t < 8; ++t) { a[t] = lo[t]; a[t + 8] = hi[t]; }
        afr[i] = a;
      }
#pragma unroll
      for (int i = 0; i < 4; ++i)
#pragma unroll
        for (int j = 0; j < 2; ++j)
          acc[i][j] = __builtin_amdgcn_wmma_f32_16x16x32_bf16(
              false, afr[i], false, bfr[j], (short)0, acc[i][j], false, false);
    }

#if USE_ASYNC_LDS
    wait_async0();        // own async loads into buf^1 landed
#else
    if (hasNext) {        // VGPR -> LDS for the next tile (other buffer)
      *(uint4*)&sA[buf ^ 1][la0] = a0;
      *(uint4*)&sA[buf ^ 1][la1] = a1;
      *(uint4*)&sB[buf ^ 1][la0] = b0;
      *(uint4*)&sB[buf ^ 1][la1] = b1;
    }
#endif
    __syncthreads();      // everyone done reading buf / writing buf^1
    buf ^= 1;
  }

  // ---- epilogue: bias (+ReLU), f32 or bf16 store ----
#pragma unroll
  for (int j = 0; j < 2; ++j) {
    const int col = bn + wc * 32 + j * 16 + lm;
    if (col >= Nc) continue;
    const float bv = bias ? bias[col] : 0.f;
#pragma unroll
    for (int i = 0; i < 4; ++i) {
      const int rbase = bm + wr * 64 + i * 16 + h * 8;  // C: VGPR g -> M=g+8*h
#pragma unroll
      for (int g = 0; g < 8; ++g) {
        float val = acc[i][j][g] + bv;
        if (RELU) val = fmaxf(val, 0.f);
        const size_t idx = (size_t)(rbase + g) * Nc + col;
        if (OUT_BF16) Cb[idx] = (__bf16)val; else Cf[idx] = val;
      }
    }
  }
}

// ---------------------------------------------------------------------------
// Edge attention: wave per (edge, head)
// ---------------------------------------------------------------------------
__global__ __launch_bounds__(256)
void edge_scores(const float* __restrict__ q, const float* __restrict__ k,
                 const int* __restrict__ ei, float* __restrict__ scores,
                 unsigned* __restrict__ menc) {
  const int task = (blockIdx.x * 256 + threadIdx.x) >> 5;   // e*H + h
  const int lane = threadIdx.x & 31;
  const int e = task >> 3, hh = task & 7;
  const int r = ei[2 * e], c = ei[2 * e + 1];
  const float* qp = q + (size_t)c * D_ + hh * DH_ + lane * 8;
  const float* kp = k + (size_t)r * D_ + hh * DH_ + lane * 8;
  const float4 q0 = *(const float4*)qp, q1 = *(const float4*)(qp + 4);
  const float4 k0 = *(const float4*)kp, k1 = *(const float4*)(kp + 4);
  float s = q0.x * k0.x + q0.y * k0.y + q0.z * k0.z + q0.w * k0.w +
            q1.x * k1.x + q1.y * k1.y + q1.z * k1.z + q1.w * k1.w;
#pragma unroll
  for (int o = 16; o > 0; o >>= 1) s += __shfl_xor(s, o, 32);
  s *= SCALE_;
  if (lane == 0) {
    scores[task] = s;
    atomicMax(&menc[c * H_ + hh], encf(s));     // segment max over dst
  }
}

__global__ __launch_bounds__(256)
void edge_exp(const int* __restrict__ ei, float* __restrict__ scores,
              const unsigned* __restrict__ menc, float* __restrict__ den) {
  const int t = blockIdx.x * 256 + threadIdx.x;
  const int e = t >> 3, hh = t & 7;
  const int c = ei[2 * e + 1];
  const float ex = expf(scores[t] - decf(menc[c * H_ + hh]));
  scores[t] = ex;
  atomicAdd(&den[c * H_ + hh], ex);             // segment sum over dst
}

__global__ __launch_bounds__(256)
void edge_agg(const int* __restrict__ ei, const float* __restrict__ scores,
              const float* __restrict__ den, const float* __restrict__ v,
              float* __restrict__ agg) {
  const int task = (blockIdx.x * 256 + threadIdx.x) >> 5;
  const int lane = threadIdx.x & 31;
  const int e = task >> 3, hh = task & 7;
  const int r = ei[2 * e], c = ei[2 * e + 1];
  const float alpha = scores[task] / den[c * H_ + hh];
  const float* vp = v + (size_t)r * D_ + hh * DH_ + lane * 8;
  float* ap = agg + (size_t)c * D_ + hh * DH_ + lane * 8;
#pragma unroll
  for (int t = 0; t < 8; ++t) atomicAdd(&ap[t], alpha * vp[t]);
}

// ---------------------------------------------------------------------------
// x = LN(xin + x2) * g + b ; emits f32 (residual chain) and bf16 (next GEMM A)
// ---------------------------------------------------------------------------
__global__ __launch_bounds__(256)
void layernorm_res(const float* __restrict__ xin, const float* __restrict__ x2,
                   const float* __restrict__ g, const float* __restrict__ b,
                   float* __restrict__ xout, __bf16* __restrict__ xbf) {
  const int row = blockIdx.x;
  const int tid = threadIdx.x;
  const size_t off = (size_t)row * D_;
  float vals[8];
  float sum = 0.f, sumsq = 0.f;
#pragma unroll
  for (int t = 0; t < 8; ++t) {
    const int i = t * 256 + tid;
    const float u = xin[off + i] + x2[off + i];
    vals[t] = u; sum += u; sumsq += u * u;
  }
#pragma unroll
  for (int o = 16; o > 0; o >>= 1) {
    sum   += __shfl_xor(sum, o, 32);
    sumsq += __shfl_xor(sumsq, o, 32);
  }
  __shared__ float ssum[8], ssq[8];
  if ((tid & 31) == 0) { ssum[tid >> 5] = sum; ssq[tid >> 5] = sumsq; }
  __syncthreads();
  sum = 0.f; sumsq = 0.f;
#pragma unroll
  for (int w = 0; w < 8; ++w) { sum += ssum[w]; sumsq += ssq[w]; }
  const float mu  = sum * (1.f / D_);
  const float var = sumsq * (1.f / D_) - mu * mu;
  const float inv = rsqrtf(var + 1e-5f);
#pragma unroll
  for (int t = 0; t < 8; ++t) {
    const int i = t * 256 + tid;
    const float y = (vals[t] - mu) * inv * g[i] + b[i];
    xout[off + i] = y;
    xbf[off + i]  = (__bf16)y;
  }
}

// ---------------------------------------------------------------------------
// Host orchestration
// ---------------------------------------------------------------------------
extern "C" void kernel_launch(void* const* d_in, const int* in_sizes, int n_in,
                              void* d_out, int out_size, void* d_ws, size_t ws_size,
                              hipStream_t stream) {
  const float* feats = (const float*)d_in[0];
  const int*   ei    = (const int*)  d_in[1];
  const float* Wq    = (const float*)d_in[2];
  const float* bq    = (const float*)d_in[3];
  const float* Wk    = (const float*)d_in[4];
  const float* bk    = (const float*)d_in[5];
  const float* Wv    = (const float*)d_in[6];
  const float* bvv   = (const float*)d_in[7];
  const float* Wo    = (const float*)d_in[8];
  const float* bo    = (const float*)d_in[9];
  const float* W1    = (const float*)d_in[10];
  const float* b1    = (const float*)d_in[11];
  const float* W2    = (const float*)d_in[12];
  const float* b2    = (const float*)d_in[13];
  const float* g1    = (const float*)d_in[14];
  const float* be1   = (const float*)d_in[15];
  const float* g2    = (const float*)d_in[16];
  const float* be2   = (const float*)d_in[17];
  const float* Wfc   = (const float*)d_in[18];
  const float* bfc   = (const float*)d_in[19];

  // workspace layout (~242 MiB)
  constexpr size_t SZ_ND = (size_t)N_ * D_ * sizeof(float);   // 32 MiB
  char* ws = (char*)d_ws;
  float*    xf32 = (float*)(ws);                       // current x (residual)
  float*    qbuf = (float*)(ws + 1 * SZ_ND);           // q, later x2
  float*    kbuf = (float*)(ws + 2 * SZ_ND);           // k, later agg
  float*    vbuf = (float*)(ws + 3 * SZ_ND);           // v
  __bf16*   hid  = (__bf16*)(ws + 4 * SZ_ND);          // N x HID bf16 (64 MiB)
  __bf16*   xbf  = (__bf16*)((char*)hid + (size_t)N_ * HID_ * 2);
  __bf16*   wbt  = (__bf16*)((char*)xbf + (size_t)N_ * D_ * 2);  // B^T (<=32MiB)
  float*    scores = (float*)((char*)wbt + (size_t)D_ * HID_ * 2);
  unsigned* menc = (unsigned*)((char*)scores + (size_t)E_ * H_ * 4);
  float*    den  = (float*)((char*)menc + (size_t)N_ * H_ * 4);

  auto cvt = [&](const float* src, __bf16* dst, size_t n) {
    const int n4 = (int)(n / 4);
    cvt_f32_bf16<<<(n4 + 255) / 256, 256, 0, stream>>>((const float4*)src, dst, n4);
  };
  auto tr = [&](const float* W, int K, int Nc) {
    dim3 g((Nc + 31) / 32, K / 32), b(32, 8);
    transpose_cvt<<<g, b, 0, stream>>>(W, wbt, K, Nc);
  };
  auto gemm_f32 = [&](const __bf16* A, const float* bias, float* C,
                      int M, int Nc, int K) {
    dim3 g((Nc + 127) / 128, M / 128);
    gemm_wmma<false, false><<<g, 256, 0, stream>>>(A, wbt, bias, C, nullptr, M, Nc, K);
  };
  auto gemm_relu_bf16 = [&](const __bf16* A, const float* bias, __bf16* C,
                            int M, int Nc, int K) {
    dim3 g((Nc + 127) / 128, M / 128);
    gemm_wmma<true, true><<<g, 256, 0, stream>>>(A, wbt, bias, nullptr, C, M, Nc, K);
  };

  (void)hipMemcpyAsync(xf32, feats, SZ_ND, hipMemcpyDeviceToDevice, stream);

  for (int l = 0; l < L_; ++l) {
    const size_t wdd = (size_t)l * D_ * D_;
    const size_t wdh = (size_t)l * D_ * HID_;
    const size_t vd  = (size_t)l * D_;
    const size_t vh  = (size_t)l * HID_;

    // q, k, v projections
    cvt(xf32, xbf, (size_t)N_ * D_);
    tr(Wq + wdd, D_, D_);  gemm_f32(xbf, bq  + vd, qbuf, N_, D_, D_);
    tr(Wk + wdd, D_, D_);  gemm_f32(xbf, bk  + vd, kbuf, N_, D_, D_);
    tr(Wv + wdd, D_, D_);  gemm_f32(xbf, bvv + vd, vbuf, N_, D_, D_);

    // edge softmax attention
    (void)hipMemsetAsync(menc, 0, (size_t)N_ * H_ * 4, stream);
    (void)hipMemsetAsync(den,  0, (size_t)N_ * H_ * 4, stream);
    edge_scores<<<E_ * H_ / 8, 256, 0, stream>>>(qbuf, kbuf, ei, scores, menc);
    edge_exp<<<E_ * H_ / 256, 256, 0, stream>>>(ei, scores, menc, den);
    (void)hipMemsetAsync(kbuf, 0, SZ_ND, stream);         // reuse kbuf as agg
    edge_agg<<<E_ * H_ / 8, 256, 0, stream>>>(ei, scores, den, vbuf, kbuf);

    // output projection + LN1
    cvt(kbuf, xbf, (size_t)N_ * D_);                      // agg -> bf16
    tr(Wo + wdd, D_, D_);  gemm_f32(xbf, bo + vd, qbuf, N_, D_, D_);   // x2
    layernorm_res<<<N_, 256, 0, stream>>>(xf32, qbuf, g1 + vd, be1 + vd, xf32, xbf);

    // FFN + LN2
    tr(W1 + wdh, D_, HID_);  gemm_relu_bf16(xbf, b1 + vh, hid, N_, HID_, D_);
    tr(W2 + wdh, HID_, D_);  gemm_f32(hid, b2 + vd, qbuf, N_, D_, HID_);
    layernorm_res<<<N_, 256, 0, stream>>>(xf32, qbuf, g2 + vd, be2 + vd, xf32, xbf);
  }

  // classifier: logits -> d_out[0 : N*NC), then final x -> d_out[N*NC : ...)
  tr(Wfc, D_, NC_);
  gemm_f32(xbf, bfc, (float*)d_out, N_, NC_, D_);
  (void)hipMemcpyAsync((float*)d_out + (size_t)N_ * NC_, xf32, SZ_ND,
                       hipMemcpyDeviceToDevice, stream);
}
